// AbsDiagNet_25211458028164
// MI455X (gfx1250) — matrix operand order repaired and checked by reading, not compiled, not run on hardware
//
#include <hip/hip_runtime.h>

typedef __attribute__((ext_vector_type(16))) _Float16 v16h;
typedef __attribute__((ext_vector_type(8)))  _Float16 v8h;
typedef __attribute__((ext_vector_type(4)))  _Float16 v4h;
typedef __attribute__((ext_vector_type(8)))  float    v8f;
typedef __attribute__((ext_vector_type(4)))  float    v4f;
typedef __attribute__((ext_vector_type(4)))  int      v4i;

constexpr int kT = 256, kB = 128, kI = 512, kH = 2048, kO = 512;
constexpr int XSTR = 520;  // LDS row stride in halfs: 512 + 8 pad (16B aligned, bank-conflict free)

#if defined(__has_builtin)
#if __has_builtin(__builtin_amdgcn_global_load_async_to_lds_b128)
#define HAVE_ASYNC 1
#endif
#endif

typedef __attribute__((address_space(1))) v4i* gptr_v4i;
typedef __attribute__((address_space(3))) v4i* lptr_v4i;

// 16-byte global->LDS copy: async DMA path on CDNA5, sync fallback otherwise.
__device__ __forceinline__ void cp16(const _Float16* g, _Float16* l) {
#ifdef HAVE_ASYNC
  __builtin_amdgcn_global_load_async_to_lds_b128((gptr_v4i)g, (lptr_v4i)l, 0, 0);
#else
  *(v8h*)l = *(const v8h*)g;
#endif
}

__device__ __forceinline__ void async_wait_le16() {
#ifdef HAVE_ASYNC
#if __has_builtin(__builtin_amdgcn_s_wait_asynccnt)
  __builtin_amdgcn_s_wait_asynccnt(16);
#else
  asm volatile("s_wait_asynccnt 16" ::: "memory");
#endif
#endif
}

__device__ __forceinline__ void async_wait_le0() {
#ifdef HAVE_ASYNC
#if __has_builtin(__builtin_amdgcn_s_wait_asynccnt)
  __builtin_amdgcn_s_wait_asynccnt(0);
#else
  asm volatile("s_wait_asynccnt 0" ::: "memory");
#endif
#endif
}

union AF { v16h v; v8h h[2]; };

__device__ __forceinline__ v8f wmma_f16(v16h a, v16h b, v8f c) {
  return __builtin_amdgcn_wmma_f32_16x16x32_f16(false, a, false, b, (short)0, c,
                                                false, false);
}

// --- Kernel 0: f32 -> f16 conversion (vectorized x4) ------------------------
__global__ void cvt_f16_kernel(const float* __restrict__ src,
                               _Float16* __restrict__ dst, int n4) {
  int i = blockIdx.x * blockDim.x + threadIdx.x;
  if (i < n4) {
    v4f x = ((const v4f*)src)[i];
    v4h y;
#pragma unroll
    for (int j = 0; j < 4; ++j) y[j] = (_Float16)x[j];
    ((v4h*)dst)[i] = y;
  }
}

// Rolled async staging of a [rows x 512] f16 tile into padded LDS.
// With 128 threads: c8 = tid&63 is iteration-invariant, row = 2j + (tid>>6).
// Keeping this loop rolled minimizes live VGPR address pairs.
__device__ __forceinline__ void stage_tile(const _Float16* __restrict__ gbase,
                                           _Float16* lbase, int tid, int iters) {
  const int c8 = tid & 63;
  const int r0 = tid >> 6;  // 0 or 1
  const _Float16* g = gbase + (size_t)r0 * kI + c8 * 8;
  _Float16* l = lbase + r0 * XSTR + c8 * 8;
#pragma unroll 1
  for (int j = 0; j < iters; ++j) {
    cp16(g, l);
    g += 2 * kI;
    l += 2 * XSTR;
  }
}

// --- Kernel 1: fused projection GEMM + abs-diag scan ------------------------
// Block: 128 threads (4 waves). Tile: B_tile=32 (rows), H_tile=64 (cols).
// Wave w: mb = w&1 selects 16-row subtile, nq = w>>1 selects 32-col group.
// State h[16x32] lives in two v8f accumulators per wave across all T steps.
// W fragments for kc<8 are hoisted into registers (loop-invariant over t),
// cutting steady-state LDS traffic from 3 to 2 ds_load_b128 per WMMA.
__global__ __launch_bounds__(128) void proj_scan_kernel(
    const _Float16* __restrict__ X16,  // [T,B,I] f16
    const _Float16* __restrict__ W16,  // [H,I]   f16
    const float* __restrict__ HH,      // [H]
    _Float16* __restrict__ Hf)         // [B,H]   f16 out
{
  __shared__ __align__(16) _Float16 Wl[64 * XSTR];      // 66,560 B
  __shared__ __align__(16) _Float16 Xl[2][32 * XSTR];   // 66,560 B

  const int tid = threadIdx.x;
  const int lane = tid & 31;
  const int wave = tid >> 5;
  const int h0 = blockIdx.x * 64;
  const int b0 = blockIdx.y * 32;
  const int mb = wave & 1;
  const int nq = wave >> 1;
  const int m = lane & 15;
  const int g = lane >> 4;

  // Stage W_IH tile (64 x 512 halfs) and X_0 tile (32 x 512 halfs) into LDS.
  stage_tile(W16 + (size_t)h0 * kI, Wl, tid, 32);
  stage_tile(X16 + (size_t)b0 * kI, &Xl[0][0], tid, 16);
  async_wait_le0();
  __syncthreads();

  const float hh0 = HH[h0 + nq * 32 + m];
  const float hh1 = HH[h0 + nq * 32 + 16 + m];
  const _Float16* Wr0 = &Wl[(nq * 32 + m) * XSTR];       // B frag col n=m
  const _Float16* Wr1 = &Wl[(nq * 32 + 16 + m) * XSTR];

  // Hoist B fragments for kc = 0..7 into registers (constant over all t).
  AF bhA[8], bhB[8];
#pragma unroll
  for (int kc = 0; kc < 8; ++kc) {
    const int k0 = kc * 32;
    bhA[kc].h[0] = *(const v8h*)(Wr0 + k0 + g * 16);
    bhA[kc].h[1] = *(const v8h*)(Wr0 + k0 + g * 16 + 8);
    bhB[kc].h[0] = *(const v8h*)(Wr1 + k0 + g * 16);
    bhB[kc].h[1] = *(const v8h*)(Wr1 + k0 + g * 16 + 8);
  }

  v8f c0 = {}, c1 = {};  // scan state (h starts at zero)

  for (int t = 0; t < kT; ++t) {
    const int cur = t & 1;
    if (t + 1 < kT) {
      // Prefetch X_{t+1} into the other buffer (async, overlaps WMMA below).
      stage_tile(X16 + (size_t)(t + 1) * (kB * kI) + (size_t)b0 * kI,
                 &Xl[cur ^ 1][0], tid, 16);
      async_wait_le16();   // batch for t complete; t+1 batch still in flight
    } else {
      async_wait_le0();
    }
    __syncthreads();

    // Seed accumulators with HH * h_prev, then accumulate X_t @ W^T on top.
    v8f a0, a1;
#pragma unroll
    for (int r = 0; r < 8; ++r) { a0[r] = c0[r] * hh0; a1[r] = c1[r] * hh1; }

    const _Float16* Ar = &Xl[cur][(mb * 16 + m) * XSTR];

    // Low half of K: B fragments already in registers; only A comes from LDS.
#pragma unroll
    for (int kc = 0; kc < 8; ++kc) {
      const int k0 = kc * 32;
      AF a;
      // A 16x32 f16 frag: lane<16 K=0..15, lane>=16 K=16..31 (contiguous, matches B)
      a.h[0] = *(const v8h*)(Ar + k0 + g * 16);
      a.h[1] = *(const v8h*)(Ar + k0 + g * 16 + 8);
      a0 = wmma_f16(a.v, bhA[kc].v, a0);
      a1 = wmma_f16(a.v, bhB[kc].v, a1);
    }
    // High half of K: stream both A and B fragments from LDS.
#pragma unroll
    for (int kc = 8; kc < 16; ++kc) {
      const int k0 = kc * 32;
      AF a, bf0, bf1;
      a.h[0] = *(const v8h*)(Ar + k0 + g * 16);
      a.h[1] = *(const v8h*)(Ar + k0 + g * 16 + 8);
      // B 32x16 f16 frag: lane<16 K=0..15, lane>=16 K=16..31 (contiguous)
      bf0.h[0] = *(const v8h*)(Wr0 + k0 + g * 16);
      bf0.h[1] = *(const v8h*)(Wr0 + k0 + g * 16 + 8);
      bf1.h[0] = *(const v8h*)(Wr1 + k0 + g * 16);
      bf1.h[1] = *(const v8h*)(Wr1 + k0 + g * 16 + 8);
      a0 = wmma_f16(a.v, bf0.v, a0);
      a1 = wmma_f16(a.v, bf1.v, a1);
    }
#pragma unroll
    for (int r = 0; r < 8; ++r) {
      c0[r] = __builtin_fabsf(a0[r]);
      c1[r] = __builtin_fabsf(a1[r]);
    }
    __syncthreads();  // everyone done reading buf `cur` before it is refilled
  }

  // Store final state as f16 for the output GEMM.
#pragma unroll
  for (int r = 0; r < 8; ++r) {
    const int row = b0 + mb * 16 + g * 8 + r;  // C/D layout: lanes>=16 -> M+8
    const int col = h0 + nq * 32 + m;
    Hf[(size_t)row * kH + col] = (_Float16)c0[r];
    Hf[(size_t)row * kH + col + 16] = (_Float16)c1[r];
  }
}

// --- Kernel 2: Y = h_final @ W_HO^T + b_HO ---------------------------------
// 128 wave-tiles of 16(b) x 32(o); 4 waves/block, 32 blocks. K = 2048.
__global__ __launch_bounds__(128) void out_gemm_kernel(
    const _Float16* __restrict__ Hf,   // [B,H] f16
    const _Float16* __restrict__ Wo,   // [O,H] f16
    const float* __restrict__ bias,    // [O]
    float* __restrict__ Y)             // [B,O] f32
{
  const int lane = threadIdx.x & 31;
  const int wave = threadIdx.x >> 5;
  const int tile = blockIdx.x * 4 + wave;  // 0..127
  const int bt = tile >> 4;                // 0..7
  const int ot = tile & 15;                // 0..15
  const int m = lane & 15, g = lane >> 4;

  const _Float16* Ar = Hf + (size_t)(bt * 16 + m) * kH;
  const _Float16* B0 = Wo + (size_t)(ot * 32 + m) * kH;
  const _Float16* B1 = B0 + (size_t)16 * kH;

  v8f acc0 = {}, acc1 = {};
#pragma unroll 4
  for (int kc = 0; kc < 64; ++kc) {
    const int k0 = kc * 32;
    AF a, bf0, bf1;
    a.h[0] = *(const v8h*)(Ar + k0 + g * 16);
    a.h[1] = *(const v8h*)(Ar + k0 + g * 16 + 8);
    bf0.h[0] = *(const v8h*)(B0 + k0 + g * 16);
    bf0.h[1] = *(const v8h*)(B0 + k0 + g * 16 + 8);
    bf1.h[0] = *(const v8h*)(B1 + k0 + g * 16);
    bf1.h[1] = *(const v8h*)(B1 + k0 + g * 16 + 8);
    acc0 = wmma_f16(a.v, bf0.v, acc0);
    acc1 = wmma_f16(a.v, bf1.v, acc1);
  }

  const float bias0 = bias[ot * 32 + m];
  const float bias1 = bias[ot * 32 + 16 + m];
#pragma unroll
  for (int r = 0; r < 8; ++r) {
    const int row = bt * 16 + g * 8 + r;
    Y[(size_t)row * kO + ot * 32 + m] = acc0[r] + bias0;
    Y[(size_t)row * kO + ot * 32 + 16 + m] = acc1[r] + bias1;
  }
}

extern "C" void kernel_launch(void* const* d_in, const int* in_sizes, int n_in,
                              void* d_out, int out_size, void* d_ws, size_t ws_size,
                              hipStream_t stream) {
  (void)in_sizes; (void)n_in; (void)out_size; (void)ws_size;
  const float* X   = (const float*)d_in[0];  // [T,B,I]
  const float* Wih = (const float*)d_in[1];  // [H,I]
  const float* HH  = (const float*)d_in[2];  // [H]
  const float* Who = (const float*)d_in[3];  // [O,H]
  const float* bHo = (const float*)d_in[4];  // [O]
  float* Y = (float*)d_out;                  // [B,O]

  // Workspace layout (f16 staging): X16 | W_IH16 | W_HO16 | Hfinal16
  char* ws = (char*)d_ws;
  _Float16* X16  = (_Float16*)(ws);
  _Float16* W16  = (_Float16*)(ws + (size_t)33554432);
  _Float16* Wo16 = (_Float16*)(ws + (size_t)33554432 + 2097152);
  _Float16* Hf16 = (_Float16*)(ws + (size_t)33554432 + 2 * 2097152);

  const int nX = kT * kB * kI, nWih = kH * kI, nWho = kO * kH;
  cvt_f16_kernel<<<(nX / 4) / 256, 256, 0, stream>>>(X, X16, nX / 4);
  cvt_f16_kernel<<<(nWih / 4) / 256, 256, 0, stream>>>(Wih, W16, nWih / 4);
  cvt_f16_kernel<<<(nWho / 4) / 256, 256, 0, stream>>>(Who, Wo16, nWho / 4);

  proj_scan_kernel<<<dim3(kH / 64, kB / 32), 128, 0, stream>>>(X16, W16, HH, Hf16);
  out_gemm_kernel<<<dim3(32), 128, 0, stream>>>(Hf16, Wo16, bHo, Y);
}
